// MultiHeadSelfAttention_49039936586428
// MI455X (gfx1250) — matrix operand order repaired and checked by reading, not compiled, not run on hardware
//
#include <hip/hip_runtime.h>
#include <hip/hip_bf16.h>

// ---------------- types ----------------
typedef __bf16 bf16;
typedef __attribute__((ext_vector_type(8)))  __bf16 v8bf;
typedef __attribute__((ext_vector_type(16))) __bf16 v16bf;
typedef __attribute__((ext_vector_type(8)))  float  v8f;

#define EMBED   512
#define NTOK    1024          // H*W
#define BATCH   8
#define HEADS   8
#define HDIM    64
#define MTOT    (BATCH * NTOK)   // 8192
#define QKVN    (3 * EMBED)      // 1536
#define ATT_SCALE 0.125f         // 1/sqrt(64)

// ---------------- helpers ----------------
__device__ __forceinline__ bf16 f2bf(float f) {
  union { float f; unsigned u; } v; v.f = f;
  unsigned r = v.u + 0x7FFFu + ((v.u >> 16) & 1u);   // round-to-nearest-even
  unsigned short h = (unsigned short)(r >> 16);
  return __builtin_bit_cast(bf16, h);
}

// WMMA wrapper: D = A(16x32 bf16) * B(32x16 bf16) + C(16x16 f32)
__device__ __forceinline__ v8f wmma_bf16(v16bf a, v16bf b, v8f c) {
  return __builtin_amdgcn_wmma_f32_16x16x32_bf16(
      /*neg_a=*/false, a, /*neg_b=*/false, b,
      /*c_mod=*/(short)0, c, /*reuse_a=*/false, /*reuse_b=*/false);
}

// A-operand (16x32, M=rows, K=cols) from row-major src (tile origin), ld in elems.
// Per ISA: lane L: row = L&15; K chunks {koff..koff+7} U {16+koff..16+koff+7}, koff=8*(L>>4).
__device__ __forceinline__ v16bf load_a(const bf16* src, int ld, int lane) {
  int r    = lane & 15;
  int koff = (lane >> 4) << 3;
  const bf16* p = src + (size_t)r * ld;
  v8bf lo = *(const v8bf*)(p + koff);
  v8bf hi = *(const v8bf*)(p + 16 + koff);
  return __builtin_shufflevector(lo, hi, 0,1,2,3,4,5,6,7,8,9,10,11,12,13,14,15);
}

// B-operand (32x16, K=rows, N=cols) from "Bt" storage: bt[n][k] row-major, ld in elems.
// Per ISA: lane L: col n = L&15; K chunks {koff..koff+7} U {16+koff..16+koff+7}, koff=8*(L>>4).
__device__ __forceinline__ v16bf load_b(const bf16* bt, int ld, int lane) {
  int n  = lane & 15;
  int kb = (lane >> 4) << 3;
  const bf16* p = bt + (size_t)n * ld;
  v8bf lo = *(const v8bf*)(p + kb);
  v8bf hi = *(const v8bf*)(p + 16 + kb);
  return __builtin_shufflevector(lo, hi, 0,1,2,3,4,5,6,7,8,9,10,11,12,13,14,15);
}

// ---------------- 1) LayerNorm (channel dim) -> bf16 xn[B*N, C] ----------------
// grid: B*(N/16) blocks, 256 threads. LDS transpose for coalesced x loads.
__global__ void ln_kernel(const float* __restrict__ x, const float* __restrict__ gamma,
                          const float* __restrict__ beta, bf16* __restrict__ xn) {
  __shared__ float tile[16][EMBED + 1];
  int blk = blockIdx.x;
  int b   = blk >> 6;                 // N/16 = 64 blocks per batch
  int n0  = (blk & 63) << 4;          // 16 tokens
  int t   = threadIdx.x;
  int nl  = t & 15, cl = t >> 4;      // cl in 0..15
  const float* xb = x + (size_t)b * EMBED * NTOK;
  for (int c0 = 0; c0 < EMBED; c0 += 16)
    tile[nl][c0 + cl] = xb[(size_t)(c0 + cl) * NTOK + n0 + nl];
  __syncthreads();

  int wave = t >> 5, lane = t & 31;   // 8 waves x 2 tokens
  for (int i = 0; i < 2; ++i) {
    int n = wave * 2 + i;
    float vals[16];
    float s = 0.f, ss = 0.f;
    #pragma unroll
    for (int j = 0; j < 16; ++j) {
      float v = tile[n][lane + 32 * j];
      vals[j] = v; s += v; ss += v * v;
    }
    #pragma unroll
    for (int off = 16; off >= 1; off >>= 1) {
      s  += __shfl_xor(s,  off, 32);
      ss += __shfl_xor(ss, off, 32);
    }
    float mu  = s * (1.0f / EMBED);
    float var = ss * (1.0f / EMBED) - mu * mu;
    float inv = rsqrtf(var + 1e-5f);
    bf16* outp = xn + ((size_t)b * NTOK + n0 + n) * EMBED;
    #pragma unroll
    for (int j = 0; j < 16; ++j) {
      int c = lane + 32 * j;
      outp[c] = f2bf((vals[j] - mu) * inv * gamma[c] + beta[c]);
    }
  }
}

// ---------------- 2) weight convert + transpose: f32 [rows][cols] -> bf16 [cols][rows] ----------------
__global__ void transpose_bf16(const float* __restrict__ w, bf16* __restrict__ wt,
                               int rows, int cols) {
  __shared__ float t[32][33];
  int bx = blockIdx.x * 32;   // col base
  int by = blockIdx.y * 32;   // row base
  int tx = threadIdx.x & 31, ty = threadIdx.x >> 5;   // 256 threads
  for (int i = ty; i < 32; i += 8)
    t[i][tx] = w[(size_t)(by + i) * cols + bx + tx];
  __syncthreads();
  for (int i = ty; i < 32; i += 8)
    wt[(size_t)(bx + i) * rows + by + tx] = f2bf(t[tx][i]);
}

// ---------------- 3) QKV GEMM: xn[8192,512] @ Wt -> scatter q,k,vT (bf16) ----------------
// block: 256 thr (8 waves), tile 128(M) x 64(N); grid (64, 24)
__global__ void qkv_gemm(const bf16* __restrict__ xn, const bf16* __restrict__ wt,
                         const float* __restrict__ bias,
                         bf16* __restrict__ q, bf16* __restrict__ k, bf16* __restrict__ vt) {
  int wave = threadIdx.x >> 5, lane = threadIdx.x & 31;
  int m0 = blockIdx.x * 128 + wave * 16;
  int n0 = blockIdx.y * 64;
  v8f acc[4] = {};
  const bf16* arow = xn + (size_t)m0 * EMBED;
  for (int k0 = 0; k0 < EMBED; k0 += 32) {
    v16bf a = load_a(arow + k0, EMBED, lane);
    #pragma unroll
    for (int t = 0; t < 4; ++t) {
      v16bf bm = load_b(wt + (size_t)(n0 + t * 16) * EMBED + k0, EMBED, lane);
      acc[t] = wmma_bf16(a, bm, acc[t]);
    }
  }
  int hi = lane >> 4, nl = lane & 15;
  #pragma unroll
  for (int t = 0; t < 4; ++t) {
    int col   = n0 + t * 16 + nl;
    int which = col >> 9;              // 0=q 1=k 2=v (uniform per tile)
    int head  = (col >> 6) & 7;
    int dpos  = col & 63;
    float bv  = bias[col];
    #pragma unroll
    for (int v = 0; v < 8; ++v) {
      int m = m0 + v + 8 * hi;
      int bb = m >> 10, n = m & 1023;
      float val = acc[t][v] + bv;
      size_t hoff = (size_t)(bb * HEADS + head);
      if (which == 0)      q [(hoff * NTOK + n) * HDIM + dpos] = f2bf(val * ATT_SCALE);
      else if (which == 1) k [(hoff * NTOK + n) * HDIM + dpos] = f2bf(val);
      else                 vt[(hoff * HDIM + dpos) * NTOK + n] = f2bf(val);
    }
  }
}

// ---------------- 4) flash attention ----------------
// grid: B*H*(N/64) = 1024 blocks, 128 thr (4 waves); wave = 16 query rows, d=64 output.
__global__ void attn_kernel(const bf16* __restrict__ q, const bf16* __restrict__ kmat,
                            const bf16* __restrict__ vt, bf16* __restrict__ o) {
  __shared__ __align__(64) bf16 pbuf[4][16][64];    // per-wave P tile (8 KB)
  int wave = threadIdx.x >> 5, lane = threadIdx.x & 31;
  int blk = blockIdx.x;
  int q0  = (blk & 15) << 6;
  int bh  = blk >> 4;                                // b*8 + h
  const bf16* qh = q    + (size_t)bh * NTOK * HDIM;
  const bf16* kh = kmat + (size_t)bh * NTOK * HDIM;
  const bf16* vh = vt   + (size_t)bh * HDIM * NTOK;
  int qrow = q0 + wave * 16;

  v16bf aq0 = load_a(qh + (size_t)qrow * HDIM + 0,  HDIM, lane);
  v16bf aq1 = load_a(qh + (size_t)qrow * HDIM + 32, HDIM, lane);

  v8f oacc[4] = {};
  float mrow[8], lrow[8];
  #pragma unroll
  for (int i = 0; i < 8; ++i) { mrow[i] = -1e30f; lrow[i] = 0.f; }
  int hi = lane >> 4, nl = lane & 15;

  for (int kb = 0; kb < NTOK; kb += 64) {
    // ---- scores: S[16 x 64] ----
    v8f s[4];
    #pragma unroll
    for (int t = 0; t < 4; ++t) {
      v16bf b0 = load_b(kh + (size_t)(kb + t * 16) * HDIM + 0,  HDIM, lane);
      v16bf b1 = load_b(kh + (size_t)(kb + t * 16) * HDIM + 32, HDIM, lane);
      v8f z = {};
      z = wmma_bf16(aq0, b0, z);
      z = wmma_bf16(aq1, b1, z);
      s[t] = z;
    }
    // ---- block row max (reduce over 4 col-tiles, then 16 lanes in half) ----
    float bm[8];
    #pragma unroll
    for (int v = 0; v < 8; ++v)
      bm[v] = fmaxf(fmaxf(s[0][v], s[1][v]), fmaxf(s[2][v], s[3][v]));
    #pragma unroll
    for (int off = 1; off < 16; off <<= 1)
      #pragma unroll
      for (int v = 0; v < 8; ++v)
        bm[v] = fmaxf(bm[v], __shfl_xor(bm[v], off, 32));
    // ---- online softmax update ----
    float prt[8];
    #pragma unroll
    for (int v = 0; v < 8; ++v) {
      float mn = fmaxf(mrow[v], bm[v]);
      float rs = __expf(mrow[v] - mn);
      mrow[v] = mn;
      lrow[v] *= rs;
      #pragma unroll
      for (int t = 0; t < 4; ++t) oacc[t][v] *= rs;
      prt[v] = 0.f;
    }
    #pragma unroll
    for (int t = 0; t < 4; ++t)
      #pragma unroll
      for (int v = 0; v < 8; ++v) {
        float p = __expf(s[t][v] - mrow[v]);
        prt[v] += p;
        pbuf[wave][v + 8 * hi][t * 16 + nl] = f2bf(p);
      }
    #pragma unroll
    for (int off = 1; off < 16; off <<= 1)
      #pragma unroll
      for (int v = 0; v < 8; ++v)
        prt[v] += __shfl_xor(prt[v], off, 32);
    #pragma unroll
    for (int v = 0; v < 8; ++v) lrow[v] += prt[v];

    __syncthreads();   // make P tile visible (D-layout -> A-layout reshape via LDS)

    // ---- O += P @ V ----
    #pragma unroll
    for (int kc = 0; kc < 2; ++kc) {
      v16bf ap = load_a(&pbuf[wave][0][kc * 32], 64, lane);
      #pragma unroll
      for (int t = 0; t < 4; ++t) {
        v16bf bv = load_b(vh + (size_t)(t * 16) * NTOK + kb + kc * 32, NTOK, lane);
        oacc[t] = wmma_bf16(ap, bv, oacc[t]);
      }
    }
    __syncthreads();
  }
  // ---- finalize: o[b*N+n][h*64+d] ----
  int b = bh >> 3, h = bh & 7;
  #pragma unroll
  for (int t = 0; t < 4; ++t)
    #pragma unroll
    for (int v = 0; v < 8; ++v) {
      int n = qrow + v + 8 * hi;
      int d = t * 16 + nl;
      float val = oacc[t][v] / lrow[v];
      o[((size_t)b * NTOK + n) * EMBED + h * HDIM + d] = f2bf(val);
    }
}

// ---------------- 5) output projection + transpose to [B,C,H,W] f32 ----------------
// grid (64, 8), 256 thr, tile 128 x 64.
__global__ void proj_gemm(const bf16* __restrict__ o, const bf16* __restrict__ wt,
                          const float* __restrict__ bias, float* __restrict__ out) {
  int wave = threadIdx.x >> 5, lane = threadIdx.x & 31;
  int m0 = blockIdx.x * 128 + wave * 16;
  int n0 = blockIdx.y * 64;
  v8f acc[4] = {};
  const bf16* arow = o + (size_t)m0 * EMBED;
  for (int k0 = 0; k0 < EMBED; k0 += 32) {
    v16bf a = load_a(arow + k0, EMBED, lane);
    #pragma unroll
    for (int t = 0; t < 4; ++t) {
      v16bf bm = load_b(wt + (size_t)(n0 + t * 16) * EMBED + k0, EMBED, lane);
      acc[t] = wmma_bf16(a, bm, acc[t]);
    }
  }
  int hi = lane >> 4, nl = lane & 15;
  #pragma unroll
  for (int t = 0; t < 4; ++t) {
    int col = n0 + t * 16 + nl;
    float bv = bias[col];
    #pragma unroll
    for (int v = 0; v < 8; ++v) {
      int m = m0 + v + 8 * hi;
      int b = m >> 10, n = m & 1023;
      out[((size_t)(b * EMBED + col)) * NTOK + n] = acc[t][v] + bv;
    }
  }
}

// ---------------- launcher ----------------
extern "C" void kernel_launch(void* const* d_in, const int* in_sizes, int n_in,
                              void* d_out, int out_size, void* d_ws, size_t ws_size,
                              hipStream_t stream) {
  const float* x     = (const float*)d_in[0];
  const float* gamma = (const float*)d_in[1];
  const float* beta  = (const float*)d_in[2];
  const float* wqkv  = (const float*)d_in[3];
  const float* bqkv  = (const float*)d_in[4];
  const float* wproj = (const float*)d_in[5];
  const float* bproj = (const float*)d_in[6];
  float* out = (float*)d_out;

  char* ws = (char*)d_ws;
  size_t off = 0;
  auto take = [&](size_t bytes) {
    size_t cur = off;
    off = (off + bytes + 255) & ~(size_t)255;
    return cur;
  };
  bf16* xn     = (bf16*)(ws + take((size_t)MTOT * EMBED * 2));
  bf16* wqkvT  = (bf16*)(ws + take((size_t)QKVN * EMBED * 2));
  bf16* wprojT = (bf16*)(ws + take((size_t)EMBED * EMBED * 2));
  bf16* qb     = (bf16*)(ws + take((size_t)MTOT * HDIM * HEADS / HEADS * 2 * HEADS / HEADS));
  // (explicit sizes below to avoid confusion)
  // re-take with clean arithmetic:
  off = 0;
  xn     = (bf16*)(ws + take((size_t)MTOT * EMBED * 2));          // 8 MB
  wqkvT  = (bf16*)(ws + take((size_t)QKVN * EMBED * 2));          // 1.5 MB
  wprojT = (bf16*)(ws + take((size_t)EMBED * EMBED * 2));         // 0.5 MB
  qb     = (bf16*)(ws + take((size_t)BATCH * HEADS * NTOK * HDIM * 2)); // 8 MB
  bf16* kb = (bf16*)(ws + take((size_t)BATCH * HEADS * NTOK * HDIM * 2)); // 8 MB
  bf16* vT = (bf16*)(ws + take((size_t)BATCH * HEADS * HDIM * NTOK * 2)); // 8 MB
  bf16* ob = (bf16*)(ws + take((size_t)MTOT * EMBED * 2));        // 8 MB

  // 1) layernorm
  ln_kernel<<<BATCH * (NTOK / 16), 256, 0, stream>>>(x, gamma, beta, xn);
  // 2) weight transpose/convert
  transpose_bf16<<<dim3(QKVN / 32, EMBED / 32), 256, 0, stream>>>(wqkv, wqkvT, EMBED, QKVN);
  transpose_bf16<<<dim3(EMBED / 32, EMBED / 32), 256, 0, stream>>>(wproj, wprojT, EMBED, EMBED);
  // 3) qkv projection
  qkv_gemm<<<dim3(MTOT / 128, QKVN / 64), 256, 0, stream>>>(xn, wqkvT, bqkv, qb, kb, vT);
  // 4) flash attention
  attn_kernel<<<BATCH * HEADS * (NTOK / 64), 128, 0, stream>>>(qb, kb, vT, ob);
  // 5) output projection (+ transpose back to [B,C,H,W])
  proj_gemm<<<dim3(MTOT / 128, EMBED / 64), 256, 0, stream>>>(ob, wprojT, bproj, out);
}
